// TransformerForSentiment_77627238908347
// MI455X (gfx1250) — matrix operand order repaired and checked by reading, not compiled, not run on hardware
//
#include <hip/hip_runtime.h>
#include <hip/hip_bf16.h>
#include <math.h>

// ---------------- problem constants ----------------
#define B_   8
#define S_   1024
#define E_   768
#define H_   12
#define HD_  64
#define L_   6
#define F_   3072
#define BS_  (B_ * S_)
#define EPS_ 1e-5f

// GEMM output modes (compile-time)
#define OUT_F32        0
#define OUT_BF16       1
#define OUT_BF16_RELU  2

// ---------------- WMMA vector types ----------------
typedef __attribute__((ext_vector_type(16))) __bf16 v16bf;
typedef __attribute__((ext_vector_type(8)))  __bf16 v8bf;
typedef __attribute__((ext_vector_type(8)))  float  v8f;
typedef int vec4i __attribute__((vector_size(16)));   // matches async-LDS builtin param

// f32 -> bf16 bits, round-to-nearest-even
__device__ __forceinline__ unsigned short f2bf(float f) {
  unsigned u = __float_as_uint(f);
  u = (u + 0x7FFFu + ((u >> 16) & 1u)) >> 16;
  return (unsigned short)u;
}
// Load a v16bf fragment from two 16B-aligned chunks of 8 bf16 each (LDS or global).
__device__ __forceinline__ v16bf frag16(const unsigned short* p0, const unsigned short* p1) {
  v8bf lo = *(const v8bf*)p0;
  v8bf hi = *(const v8bf*)p1;
  return __builtin_shufflevector(lo, hi, 0,1,2,3,4,5,6,7,8,9,10,11,12,13,14,15);
}

// ---------------- CDNA5 async global->LDS copy (16B), guarded ----------------
#if defined(__gfx1250__) && __has_builtin(__builtin_amdgcn_global_load_async_to_lds_b128)
#define ASYNC_LDS 1
#endif

__device__ __forceinline__ void cp16(unsigned short* lds, const unsigned short* g) {
#ifdef ASYNC_LDS
  __builtin_amdgcn_global_load_async_to_lds_b128(
      (__attribute__((address_space(1))) vec4i*)(unsigned short*)g,
      (__attribute__((address_space(3))) vec4i*)lds, 0, 0);
#else
  *(uint4*)lds = *(const uint4*)g;
#endif
}
__device__ __forceinline__ void wait_async0() {
#ifdef ASYNC_LDS
#if __has_builtin(__builtin_amdgcn_s_wait_asynccnt)
  __builtin_amdgcn_s_wait_asynccnt(0);
#else
  asm volatile("s_wait_asynccnt 0" ::: "memory");
#endif
#endif
}

// =====================================================================
// Weight transpose + f32->bf16 convert:  out[n*K + k] = bf16(in[k*N + n])
// 32x32 tiles via LDS, block 256 (32x8).
// =====================================================================
__global__ __launch_bounds__(256)
void k_transpose_bf16(const float* __restrict__ in, unsigned short* __restrict__ out,
                      int K, int N) {
  __shared__ float t[32][33];
  const int k0 = blockIdx.y * 32, n0 = blockIdx.x * 32;
  const int tx = threadIdx.x & 31, ty = threadIdx.x >> 5;
#pragma unroll
  for (int i = 0; i < 4; ++i) {
    int r = ty + i * 8;
    t[r][tx] = in[(size_t)(k0 + r) * N + n0 + tx];
  }
  __syncthreads();
#pragma unroll
  for (int i = 0; i < 4; ++i) {
    int r = ty + i * 8;                       // n-row within tile
    out[(size_t)(n0 + r) * K + k0 + tx] = f2bf(t[tx][r]);
  }
}

// =====================================================================
// Embedding + PE, dual f32 + bf16 outputs
// =====================================================================
__global__ __launch_bounds__(256)
void k_embed(const int* __restrict__ ids, const float* __restrict__ emb,
             const float* __restrict__ pe, float* __restrict__ X,
             unsigned short* __restrict__ Xb) {
  size_t i = (size_t)blockIdx.x * 256 + threadIdx.x;   // over BS_*E_ (divisible)
  int tok = (int)(i / E_);
  int e   = (int)(i % E_);
  int s   = tok % S_;
  float v = emb[(size_t)ids[tok] * E_ + e] + pe[(size_t)s * E_ + e];
  X[i]  = v;
  Xb[i] = f2bf(v);
}

// =====================================================================
// GEMM: C[M,N] = act(A[M,K] @ W[K,N] + bias[N]),  bf16 inputs, f32 accum.
// A: M x K bf16 row-major.  Wt: N x K bf16 (pre-transposed weight).
// Block 256 = 8 waves, macro tile 128x128, K-step 32.
// Wave (wm=wave&3, wn=wave>>2) computes 32x64 = 2x4 tiles -> 8 WMMAs/step.
// MODE is a template parameter -> branch-free epilogue.
// =====================================================================
template <int MODE>
__global__ __launch_bounds__(256)
void k_gemm_bf16(const unsigned short* __restrict__ A,
                 const unsigned short* __restrict__ Wt,
                 const float* __restrict__ bias, void* __restrict__ Cout,
                 int M, int N, int K) {
  __shared__ __align__(16) unsigned short Asm[128 * 40]; // A tile rows x 32k (pad 40)
  __shared__ __align__(16) unsigned short Bsm[128 * 40]; // Wt tile rows(n) x 32k

  const int tid  = threadIdx.x;
  const int wave = tid >> 5, lane = tid & 31;
  const int half = lane >> 4, ln = lane & 15;
  const int wm = wave & 3, wn = wave >> 2;
  const int m0 = blockIdx.y * 128, n0 = blockIdx.x * 128;

  v8f acc[2][4];
#pragma unroll
  for (int i = 0; i < 2; ++i)
#pragma unroll
    for (int j = 0; j < 4; ++j)
#pragma unroll
      for (int r = 0; r < 8; ++r) acc[i][j][r] = 0.f;

  for (int k0 = 0; k0 < K; k0 += 32) {
    // ---- stage A tile (128x32) and B tile (128n x 32k): 2+2 b128 per thread ----
#pragma unroll
    for (int i = 0; i < 2; ++i) {
      int f = tid + i * 256;              // 0..511
      int r = f >> 2, c = (f & 3) * 8;    // row 0..127, col 0/8/16/24
      cp16(&Asm[r * 40 + c], &A [(size_t)(m0 + r) * K + k0 + c]);
      cp16(&Bsm[r * 40 + c], &Wt[(size_t)(n0 + r) * K + k0 + c]);
    }
    wait_async0();
    __syncthreads();

    // ---- fragments ----
    v16bf aF[2], bF[4];
#pragma unroll
    for (int sm = 0; sm < 2; ++sm) {
      int row = wm * 32 + sm * 16 + ln;
      aF[sm] = frag16(&Asm[row * 40 + half * 8], &Asm[row * 40 + 16 + half * 8]);
    }
#pragma unroll
    for (int sn = 0; sn < 4; ++sn) {
      int col = wn * 64 + sn * 16 + ln;
      bF[sn] = frag16(&Bsm[col * 40 + half * 8], &Bsm[col * 40 + 16 + half * 8]);
    }
    // ---- 8 WMMAs ----
#pragma unroll
    for (int sm = 0; sm < 2; ++sm)
#pragma unroll
      for (int sn = 0; sn < 4; ++sn)
        acc[sm][sn] = __builtin_amdgcn_wmma_f32_16x16x32_bf16(
            false, aF[sm], false, bF[sn], (short)0, acc[sm][sn], false, false);
    __syncthreads();
  }

  // ---- branch-free epilogue ----
#pragma unroll
  for (int sm = 0; sm < 2; ++sm)
#pragma unroll
    for (int sn = 0; sn < 4; ++sn) {
      int col = n0 + wn * 64 + sn * 16 + ln;
      float bv = bias[col];
#pragma unroll
      for (int r = 0; r < 8; ++r) {
        int row = m0 + wm * 32 + sm * 16 + half * 8 + r;
        float v = acc[sm][sn][r] + bv;
        if (MODE == OUT_F32) {
          ((float*)Cout)[(size_t)row * N + col] = v;
        } else if (MODE == OUT_BF16) {
          ((unsigned short*)Cout)[(size_t)row * N + col] = f2bf(v);
        } else {
          ((unsigned short*)Cout)[(size_t)row * N + col] = f2bf(fmaxf(v, 0.f));
        }
      }
    }
}

// =====================================================================
// Flash attention per (b,h), bf16 Q/K/V in, bf16 O out.
// Block 128 = 4 waves; each wave owns 16 query rows; streams 32-key
// tiles with online softmax. All matmuls WMMA bf16/f32.
// =====================================================================
__global__ __launch_bounds__(128)
void k_flash_attn(const unsigned short* __restrict__ Q,
                  const unsigned short* __restrict__ Km,
                  const unsigned short* __restrict__ Vm,
                  const int* __restrict__ mask,
                  unsigned short* __restrict__ O) {
  __shared__ __align__(16) unsigned short Ksm[32 * 72];     // K tile, token-major (pad 72)
  __shared__ __align__(16) unsigned short Vt [64 * 40];     // V tile transposed (d x key)
  __shared__ __align__(16) unsigned short Psm[4][16 * 40];  // per-wave P scratch

  const int tid  = threadIdx.x;
  const int wave = tid >> 5, lane = tid & 31;
  const int half = lane >> 4, ln = lane & 15;
  const int b = blockIdx.x / H_, h = blockIdx.x % H_;
  const int q0   = blockIdx.y * 64 + wave * 16;
  const int colq = h * HD_;

  // ---- Q fragments straight from global bf16 (rows = ln) ----
  v16bf aQ[2];
  {
    const unsigned short* qp = &Q[(size_t)(b * S_ + q0 + ln) * E_ + colq];
#pragma unroll
    for (int c = 0; c < 2; ++c)
      aQ[c] = frag16(qp + c * 32 + half * 8, qp + c * 32 + 16 + half * 8);
  }

  float mrow[8], lrow[8];
  v8f accO[4];
#pragma unroll
  for (int r = 0; r < 8; ++r) { mrow[r] = -1e30f; lrow[r] = 0.f; }
#pragma unroll
  for (int g = 0; g < 4; ++g)
#pragma unroll
    for (int r = 0; r < 8; ++r) accO[g][r] = 0.f;

  for (int kt = 0; kt < S_ / 32; ++kt) {
    const int kbase = kt * 32;
    // ---- stage K tile (verbatim bf16 copy, async path) ----
#pragma unroll
    for (int i = 0; i < 2; ++i) {
      int f = tid + i * 128;              // 0..255
      int r = f >> 3, c = (f & 7) * 8;    // key 0..31, d 0/8/../56
      cp16(&Ksm[r * 72 + c], &Km[(size_t)(b * S_ + kbase + r) * E_ + colq + c]);
    }
    // ---- stage V tile transposed (unpack b128 -> 8 scattered u16) ----
#pragma unroll
    for (int i = 0; i < 2; ++i) {
      int f = tid + i * 128;
      int r = f >> 3, c = (f & 7) * 8;
      uint4 vv = *(const uint4*)&Vm[(size_t)(b * S_ + kbase + r) * E_ + colq + c];
      unsigned u0 = vv.x, u1 = vv.y, u2 = vv.z, u3 = vv.w;
      Vt[(c + 0) * 40 + r] = (unsigned short)(u0 & 0xffffu);
      Vt[(c + 1) * 40 + r] = (unsigned short)(u0 >> 16);
      Vt[(c + 2) * 40 + r] = (unsigned short)(u1 & 0xffffu);
      Vt[(c + 3) * 40 + r] = (unsigned short)(u1 >> 16);
      Vt[(c + 4) * 40 + r] = (unsigned short)(u2 & 0xffffu);
      Vt[(c + 5) * 40 + r] = (unsigned short)(u2 >> 16);
      Vt[(c + 6) * 40 + r] = (unsigned short)(u3 & 0xffffu);
      Vt[(c + 7) * 40 + r] = (unsigned short)(u3 >> 16);
    }
    wait_async0();
    __syncthreads();

    // ---- scores: 16x32 per wave = two 16x16 accs, HD=64 split as 2x32 ----
    v8f sA[2];
#pragma unroll
    for (int nh = 0; nh < 2; ++nh) {
#pragma unroll
      for (int r = 0; r < 8; ++r) sA[nh][r] = 0.f;
#pragma unroll
      for (int c = 0; c < 2; ++c) {
        const unsigned short* kp = &Ksm[(nh * 16 + ln) * 72 + c * 32 + half * 8];
        v16bf bK = frag16(kp, kp + 16);
        sA[nh] = __builtin_amdgcn_wmma_f32_16x16x32_bf16(
            false, aQ[c], false, bK, (short)0, sA[nh], false, false);
      }
    }

    const int mv0 = mask[b * S_ + kbase + ln];
    const int mv1 = mask[b * S_ + kbase + 16 + ln];
    // ---- online softmax, row = half*8 + r ----
#pragma unroll
    for (int r = 0; r < 8; ++r) {
      float s0 = mv0 ? sA[0][r] * 0.125f : -1e30f;
      float s1 = mv1 ? sA[1][r] * 0.125f : -1e30f;
      float tmax = fmaxf(s0, s1);
#pragma unroll
      for (int off = 8; off; off >>= 1) tmax = fmaxf(tmax, __shfl_xor(tmax, off, 16));
      float nm = fmaxf(mrow[r], tmax);
      float al = __expf(mrow[r] - nm);
      float p0 = __expf(s0 - nm), p1 = __expf(s1 - nm);
      float rs = p0 + p1;
#pragma unroll
      for (int off = 8; off; off >>= 1) rs += __shfl_xor(rs, off, 16);
      lrow[r] = lrow[r] * al + rs;
      mrow[r] = nm;
      accO[0][r] *= al; accO[1][r] *= al; accO[2][r] *= al; accO[3][r] *= al;
      int prow = half * 8 + r;
      Psm[wave][prow * 40 + ln]      = f2bf(p0);
      Psm[wave][prow * 40 + 16 + ln] = f2bf(p1);
    }

    // ---- P @ V ----
    v16bf aP = frag16(&Psm[wave][ln * 40 + half * 8],
                      &Psm[wave][ln * 40 + 16 + half * 8]);
#pragma unroll
    for (int g = 0; g < 4; ++g) {
      const unsigned short* vp = &Vt[(g * 16 + ln) * 40 + half * 8];
      v16bf bV = frag16(vp, vp + 16);
      accO[g] = __builtin_amdgcn_wmma_f32_16x16x32_bf16(
          false, aP, false, bV, (short)0, accO[g], false, false);
    }
    __syncthreads();
  }

  // ---- normalize + write bf16 ----
#pragma unroll
  for (int g = 0; g < 4; ++g)
#pragma unroll
    for (int r = 0; r < 8; ++r) {
      int row = b * S_ + q0 + half * 8 + r;
      O[(size_t)row * E_ + colq + g * 16 + ln] = f2bf(accO[g][r] / lrow[r]);
    }
}

// =====================================================================
// Residual + LayerNorm, dual f32 + bf16 outputs
// =====================================================================
__global__ __launch_bounds__(256)
void k_add_ln(const float* __restrict__ X, const float* __restrict__ Y,
              const float* __restrict__ g, const float* __restrict__ bb,
              float* __restrict__ OutF, unsigned short* __restrict__ OutB) {
  __shared__ float red[9];
  const int t = blockIdx.x, tid = threadIdx.x;
  const int wave = tid >> 5, lane = tid & 31;
  float v[3];
  float s = 0.f;
#pragma unroll
  for (int i = 0; i < 3; ++i) {
    int c = i * 256 + tid;
    v[i] = X[(size_t)t * E_ + c] + Y[(size_t)t * E_ + c];
    s += v[i];
  }
#pragma unroll
  for (int off = 16; off; off >>= 1) s += __shfl_xor(s, off, 32);
  if (lane == 0) red[wave] = s;
  __syncthreads();
  if (tid == 0) { float a = 0.f; for (int i = 0; i < 8; ++i) a += red[i]; red[8] = a; }
  __syncthreads();
  const float mu = red[8] * (1.0f / E_);
  float q = 0.f;
#pragma unroll
  for (int i = 0; i < 3; ++i) { float d = v[i] - mu; q += d * d; }
#pragma unroll
  for (int off = 16; off; off >>= 1) q += __shfl_xor(q, off, 32);
  __syncthreads();
  if (lane == 0) red[wave] = q;
  __syncthreads();
  if (tid == 0) { float a = 0.f; for (int i = 0; i < 8; ++i) a += red[i]; red[8] = a; }
  __syncthreads();
  const float inv = rsqrtf(red[8] * (1.0f / E_) + EPS_);
#pragma unroll
  for (int i = 0; i < 3; ++i) {
    int c = i * 256 + tid;
    float o = (v[i] - mu) * inv * g[c] + bb[c];
    OutF[(size_t)t * E_ + c] = o;
    OutB[(size_t)t * E_ + c] = f2bf(o);
  }
}

// =====================================================================
// Mean pool + tiny classifier
// =====================================================================
__global__ __launch_bounds__(256)
void k_pool(const float* __restrict__ X, float* __restrict__ P) {
  int t = blockIdx.x * 256 + threadIdx.x;   // 0 .. B_*E_-1
  int b = t / E_, e = t % E_;
  float s = 0.f;
  for (int sr = 0; sr < S_; ++sr) s += X[(size_t)(b * S_ + sr) * E_ + e];
  P[t] = s * (1.0f / S_);
}

__global__ __launch_bounds__(64)
void k_cls(const float* __restrict__ P, const float* __restrict__ Wc,
           const float* __restrict__ bc, float* __restrict__ out) {
  int t = threadIdx.x;
  if (t < B_ * 2) {
    int b = t >> 1, c = t & 1;
    float s = bc[c];
    for (int e = 0; e < E_; ++e) s += P[b * E_ + e] * Wc[e * 2 + c];
    out[b * 2 + c] = s;
  }
}

// =====================================================================
// Host orchestration
// =====================================================================
extern "C" void kernel_launch(void* const* d_in, const int* in_sizes, int n_in,
                              void* d_out, int out_size, void* d_ws, size_t ws_size,
                              hipStream_t stream) {
  const int*   ids   = (const int*)  d_in[0];
  const int*   amask = (const int*)  d_in[1];
  const float* emb   = (const float*)d_in[2];
  const float* pe    = (const float*)d_in[3];
  const float* Wq    = (const float*)d_in[4];
  const float* bq    = (const float*)d_in[5];
  const float* Wk    = (const float*)d_in[6];
  const float* bk    = (const float*)d_in[7];
  const float* Wv    = (const float*)d_in[8];
  const float* bv    = (const float*)d_in[9];
  const float* Wo    = (const float*)d_in[10];
  const float* bo    = (const float*)d_in[11];
  const float* g1    = (const float*)d_in[12];
  const float* be1   = (const float*)d_in[13];
  const float* W1    = (const float*)d_in[14];
  const float* b1    = (const float*)d_in[15];
  const float* W2    = (const float*)d_in[16];
  const float* b2    = (const float*)d_in[17];
  const float* g2    = (const float*)d_in[18];
  const float* be2   = (const float*)d_in[19];
  const float* Wc    = (const float*)d_in[20];
  const float* bc    = (const float*)d_in[21];
  float* out = (float*)d_out;

  // ---- workspace carve ----
  char* base = (char*)d_ws;
  size_t off = 0;
  auto alloc = [&](size_t bytes) -> void* {
    void* p = base + off;
    off += (bytes + 255) & ~(size_t)255;
    return p;
  };
  const size_t BSE = (size_t)BS_ * E_;
  const size_t EE  = (size_t)E_ * E_;
  const size_t EF  = (size_t)E_ * F_;
  float* X    = (float*)alloc(BSE * 4);
  float* X2f  = (float*)alloc(BSE * 4);
  float* Yf   = (float*)alloc(BSE * 4);
  float* pool = (float*)alloc((size_t)B_ * E_ * 4);
  unsigned short* Xb  = (unsigned short*)alloc(BSE * 2);
  unsigned short* X2b = (unsigned short*)alloc(BSE * 2);
  unsigned short* Qb  = (unsigned short*)alloc(BSE * 2);
  unsigned short* Kb  = (unsigned short*)alloc(BSE * 2);
  unsigned short* Vb  = (unsigned short*)alloc(BSE * 2);
  unsigned short* Ab  = (unsigned short*)alloc(BSE * 2);
  unsigned short* Hb  = (unsigned short*)alloc((size_t)BS_ * F_ * 2);
  const size_t WPL = 4 * EE + 2 * EF;                    // bf16 elems per layer
  unsigned short* WT = (unsigned short*)alloc(WPL * L_ * 2);

  // ---- one-time (per forward) weight transpose-convert to bf16 NxK ----
  const dim3 tEE(E_ / 32, E_ / 32);
  const dim3 tEF(F_ / 32, E_ / 32);   // W1: K=E rows, N=F cols
  const dim3 tFE(E_ / 32, F_ / 32);   // W2: K=F rows, N=E cols
  for (int l = 0; l < L_; ++l) {
    unsigned short* wt = WT + (size_t)l * WPL;
    k_transpose_bf16<<<tEE, 256, 0, stream>>>(Wq + l * EE, wt + 0 * EE, E_, E_);
    k_transpose_bf16<<<tEE, 256, 0, stream>>>(Wk + l * EE, wt + 1 * EE, E_, E_);
    k_transpose_bf16<<<tEE, 256, 0, stream>>>(Wv + l * EE, wt + 2 * EE, E_, E_);
    k_transpose_bf16<<<tEE, 256, 0, stream>>>(Wo + l * EE, wt + 3 * EE, E_, E_);
    k_transpose_bf16<<<tEF, 256, 0, stream>>>(W1 + l * EF, wt + 4 * EE, E_, F_);
    k_transpose_bf16<<<tFE, 256, 0, stream>>>(W2 + l * EF, wt + 4 * EE + EF, F_, E_);
  }

  // ---- embedding ----
  k_embed<<<(BS_ * E_) / 256, 256, 0, stream>>>(ids, emb, pe, X, Xb);

  const dim3 gE(E_ / 128, BS_ / 128);
  const dim3 gF(F_ / 128, BS_ / 128);
  const dim3 gA(B_ * H_, S_ / 64);

  for (int l = 0; l < L_; ++l) {
    unsigned short* wt = WT + (size_t)l * WPL;
    const unsigned short* wqT = wt + 0 * EE;
    const unsigned short* wkT = wt + 1 * EE;
    const unsigned short* wvT = wt + 2 * EE;
    const unsigned short* woT = wt + 3 * EE;
    const unsigned short* w1T = wt + 4 * EE;
    const unsigned short* w2T = wt + 4 * EE + EF;
    const float* bql = bq + (size_t)l * E_;  const float* bkl = bk + (size_t)l * E_;
    const float* bvl = bv + (size_t)l * E_;  const float* bol = bo + (size_t)l * E_;
    const float* b1l = b1 + (size_t)l * F_;  const float* b2l = b2 + (size_t)l * E_;
    const float* g1l = g1 + (size_t)l * E_;  const float* be1l = be1 + (size_t)l * E_;
    const float* g2l = g2 + (size_t)l * E_;  const float* be2l = be2 + (size_t)l * E_;

    k_gemm_bf16<OUT_BF16><<<gE, 256, 0, stream>>>(Xb, wqT, bql, Qb, BS_, E_, E_);
    k_gemm_bf16<OUT_BF16><<<gE, 256, 0, stream>>>(Xb, wkT, bkl, Kb, BS_, E_, E_);
    k_gemm_bf16<OUT_BF16><<<gE, 256, 0, stream>>>(Xb, wvT, bvl, Vb, BS_, E_, E_);
    k_flash_attn<<<gA, 128, 0, stream>>>(Qb, Kb, Vb, amask, Ab);
    k_gemm_bf16<OUT_F32><<<gE, 256, 0, stream>>>(Ab, woT, bol, Yf, BS_, E_, E_);
    k_add_ln<<<BS_, 256, 0, stream>>>(X, Yf, g1l, be1l, X2f, X2b);
    k_gemm_bf16<OUT_BF16_RELU><<<gF, 256, 0, stream>>>(X2b, w1T, b1l, Hb, BS_, F_, E_);
    k_gemm_bf16<OUT_F32><<<gE, 256, 0, stream>>>(Hb, w2T, b2l, Yf, BS_, E_, F_);
    k_add_ln<<<BS_, 256, 0, stream>>>(X2f, Yf, g2l, be2l, X, Xb);
  }

  k_pool<<<(B_ * E_) / 256, 256, 0, stream>>>(X, pool);
  k_cls<<<1, 64, 0, stream>>>(pool, Wc, bc, out);
}
